// MobileAttentionBlock_39960375722355
// MI455X (gfx1250) — matrix operand order repaired and checked by reading, not compiled, not run on hardware
//
#include <hip/hip_runtime.h>
#include <hip/hip_bf16.h>

// MI455X / gfx1250: wave32, WMMA f16 16x16x32 with f32 accum.
// Pipeline: rmsnorm -> (conv_kv, weight-prep) -> 3x WMMA GEMM (Q,K,V)
//           -> flash attention (WMMA QK^T + PV, online softmax)
//           -> WMMA GEMM out-proj with bias/gamma/residual epilogue.
// gfx1250 async data movement for tile staging:
//   - GLOBAL_LOAD_ASYNC_TO_LDS_B128 (ASYNCcnt) for GEMM A tiles / attn K tiles
//   - TDM tensor_load_to_lds (TENSORcnt) for the attention K tile
// __launch_bounds__(256,1) on the WMMA kernels to avoid VGPR spills seen in
// round-3 asm (scratch_load_b128 reloads inside the flash-attention loop).

typedef _Float16 half_t;
typedef __attribute__((ext_vector_type(16))) _Float16 v16h;
typedef __attribute__((ext_vector_type(8)))  _Float16 v8h;
typedef __attribute__((ext_vector_type(8)))  float    v8f;

#define NB   8
#define NH   32
#define NW   32
#define NC   512
#define NS   1024
#define HEADS 8
#define KD   64
#define BHW  8192

#if defined(__has_builtin)
#if __has_builtin(__builtin_amdgcn_global_load_async_to_lds_b128) && \
    __has_builtin(__builtin_amdgcn_s_wait_asynccnt)
#define USE_ASYNC_LDS 1
#endif
#if __has_builtin(__builtin_amdgcn_tensor_load_to_lds) && \
    __has_builtin(__builtin_amdgcn_s_wait_tensorcnt)
#define USE_TDM 1
#endif
#endif

typedef unsigned int u32x4 __attribute__((ext_vector_type(4)));
typedef int          i32x4 __attribute__((ext_vector_type(4)));
typedef int          i32x8 __attribute__((ext_vector_type(8)));
typedef int          vsi4  __attribute__((vector_size(16)));   // int4 (gcc-style)
#define AS1 __attribute__((address_space(1)))
#define AS3 __attribute__((address_space(3)))

// 16-byte global -> LDS copy, async where available.
__device__ __forceinline__ void copy16_to_lds(half_t* lds, const half_t* g) {
#ifdef USE_ASYNC_LDS
  __builtin_amdgcn_global_load_async_to_lds_b128(
      (AS1 vsi4*)g, (AS3 vsi4*)lds, 0, 0);
#else
  *(v8h*)lds = *(const v8h*)g;
#endif
}
__device__ __forceinline__ void async_join() {
#ifdef USE_ASYNC_LDS
  __builtin_amdgcn_s_wait_asynccnt(0);
#endif
}

// ---------------------------------------------------------------- RMSNorm
__global__ void rmsnorm_kernel(const float* __restrict__ in,
                               const float* __restrict__ scale,
                               half_t* __restrict__ xo) {
  const int wid  = threadIdx.x >> 5;
  const int lane = threadIdx.x & 31;
  const int pix  = blockIdx.x * 8 + wid;
  const float* row = in + (size_t)pix * NC;
  float vals[16];
  float ss = 0.f;
#pragma unroll
  for (int i = 0; i < 16; ++i) {
    float v = row[lane + i * 32];
    vals[i] = v;
    ss += v * v;
  }
#pragma unroll
  for (int m = 16; m >= 1; m >>= 1) ss += __shfl_xor(ss, m, 32);
  const float r = rsqrtf(ss * (1.0f / NC) + 1e-6f);
  half_t* orow = xo + (size_t)pix * NC;
#pragma unroll
  for (int i = 0; i < 16; ++i) {
    int c = lane + i * 32;
    orow[c] = (half_t)(vals[i] * r * scale[c]);
  }
}

// ------------------------------------------------------- grouped 3x3 convs
__global__ void conv_kv_kernel(const half_t* __restrict__ x,
                               const float* __restrict__ kw,
                               const float* __restrict__ vw,
                               half_t* __restrict__ ko, half_t* __restrict__ vo) {
  const int idx = blockIdx.x * blockDim.x + threadIdx.x;   // BHW*64
  const int o   = idx & 63;
  const int pix = idx >> 6;
  const int px  = pix & 31, py = (pix >> 5) & 31, pb = pix >> 10;
  float ak = 0.f, av = 0.f;
  for (int dy = 0; dy < 3; ++dy) {
    const int yy = py + dy - 1;
    if (yy < 0 || yy >= NH) continue;
    for (int dx = 0; dx < 3; ++dx) {
      const int xx = px + dx - 1;
      if (xx < 0 || xx >= NW) continue;
      const half_t* xp  = x + (((size_t)(pb * NH + yy) * NW + xx) * NC) + o * 8;
      const float*  kwp = kw + ((dy * 3 + dx) * 8) * 64 + o;
      const float*  vwp = vw + ((dy * 3 + dx) * 8) * 64 + o;
#pragma unroll
      for (int ci = 0; ci < 8; ++ci) {
        float xv = (float)xp[ci];
        ak += xv * kwp[ci * 64];
        av += xv * vwp[ci * 64];
      }
    }
  }
  ko[idx] = (half_t)ak;
  vo[idx] = (half_t)av;
}

// ------------------------------------------- weight prep: Wcomb = q_w @ Wq
__global__ void wcomb_kernel(const float* __restrict__ qw,
                             const float* __restrict__ Wq,
                             half_t* __restrict__ out) {
  const int idx = blockIdx.x * blockDim.x + threadIdx.x;   // 512*512
  const int o = idx & 511, c = idx >> 9;
  float acc = 0.f;
  for (int m = 0; m < 512; ++m) acc += qw[c * 512 + m] * Wq[m * 512 + o];
  out[idx] = (half_t)acc;
}

// ---------------- weight prep: fold repeat(.,8) into Wk/Wv -> [64 x 512]
__global__ void wred_kernel(const float* __restrict__ Wk,
                            const float* __restrict__ Wv,
                            half_t* __restrict__ ok, half_t* __restrict__ ov) {
  const int idx = blockIdx.x * blockDim.x + threadIdx.x;   // 64*512
  const int o = idx & 511, c = idx >> 9;
  float ak = 0.f, av = 0.f;
#pragma unroll
  for (int r = 0; r < 8; ++r) {
    ak += Wk[(c * 8 + r) * 512 + o];
    av += Wv[(c * 8 + r) * 512 + o];
  }
  ok[idx] = (half_t)ak;
  ov[idx] = (half_t)av;
}

__global__ void cvt16_kernel(const float* __restrict__ in,
                             half_t* __restrict__ out, int n) {
  const int i = blockIdx.x * blockDim.x + threadIdx.x;
  if (i < n) out[i] = (half_t)in[i];
}

// --------------------------------------------------------- WMMA GEMM
// C[M x 512] = A[M x Ka](f16) @ Bw[Ka x 512](f16) + bias
// MODE 0: write f16.  MODE 1: f32 out = acc*gamma + residual.
template <int MODE>
__global__ __launch_bounds__(256, 1)
void gemm_kernel(const half_t* __restrict__ A,
                 const half_t* __restrict__ Bw,
                 const float* __restrict__ bias, int Ka,
                 half_t* __restrict__ outh,
                 float* __restrict__ outf,
                 const float* __restrict__ gamma,
                 const float* __restrict__ resid) {
  __shared__ half_t At[128 * 32];
  __shared__ half_t Bt[128 * 32];      // transposed [n][k]
  const int t = threadIdx.x;
  const int lane = t & 31, wid = t >> 5;
  const int wm = wid >> 1, wn = wid & 1;
  const int hh = lane >> 4, l15 = lane & 15;
  const int bm = blockIdx.y * 128, bn = blockIdx.x * 128;

  v8f acc[2][4];
  v8f zz = {};
#pragma unroll
  for (int i = 0; i < 2; ++i)
#pragma unroll
    for (int j = 0; j < 4; ++j) acc[i][j] = zz;

  const int kiters = Ka >> 5;
  for (int kt = 0; kt < kiters; ++kt) {
    const int k0 = kt * 32;
    __syncthreads();
#pragma unroll
    for (int c = 0; c < 2; ++c) {               // A tile via async-to-LDS
      int id = t + c * 256;
      int row = id >> 2, c8 = (id & 3) * 8;
      copy16_to_lds(&At[row * 32 + c8],
                    &A[(size_t)(bm + row) * Ka + k0 + c8]);
    }
#pragma unroll
    for (int c = 0; c < 2; ++c) {               // B tile transposed in LDS
      int id = t + c * 256;
      int k = id >> 4, n8 = (id & 15) * 8;
      v8h bvv = *(const v8h*)&Bw[(size_t)(k0 + k) * 512 + bn + n8];
#pragma unroll
      for (int j = 0; j < 8; ++j) Bt[(n8 + j) * 32 + k] = bvv[j];
    }
    async_join();
    __syncthreads();

    v16h afr[2], bfr[4];
#pragma unroll
    for (int ms = 0; ms < 2; ++ms) {            // A frag: interleaved K
      int m = wm * 32 + ms * 16 + l15;
      union { v16h v; v8h h2[2]; } u;
      u.h2[0] = *(const v8h*)&At[m * 32 + hh * 8];
      u.h2[1] = *(const v8h*)&At[m * 32 + 16 + hh * 8];
      afr[ms] = u.v;
    }
#pragma unroll
    for (int ns = 0; ns < 4; ++ns) {            // B frag: interleaved K
      int n = wn * 64 + ns * 16 + l15;
      union { v16h v; v8h h2[2]; } u;
      u.h2[0] = *(const v8h*)&Bt[n * 32 + hh * 8];
      u.h2[1] = *(const v8h*)&Bt[n * 32 + 16 + hh * 8];
      bfr[ns] = u.v;
    }
#pragma unroll
    for (int ms = 0; ms < 2; ++ms)
#pragma unroll
      for (int ns = 0; ns < 4; ++ns)
        acc[ms][ns] = __builtin_amdgcn_wmma_f32_16x16x32_f16(
            false, afr[ms], false, bfr[ns], (short)0, acc[ms][ns], false, false);
  }

#pragma unroll
  for (int ms = 0; ms < 2; ++ms) {
#pragma unroll
    for (int ns = 0; ns < 4; ++ns) {
      const int col = bn + wn * 64 + ns * 16 + l15;
      const float bv = bias[col];
#pragma unroll
      for (int r = 0; r < 8; ++r) {
        const int row = bm + wm * 32 + ms * 16 + r + 8 * hh;
        const float v = acc[ms][ns][r] + bv;
        if (MODE == 0) {
          outh[(size_t)row * 512 + col] = (half_t)v;
        } else {
          outf[(size_t)row * 512 + col] =
              v * gamma[col] + resid[(size_t)row * 512 + col];
        }
      }
    }
  }
}

// ------------------------------------------------- flash attention (WMMA)
__global__ __launch_bounds__(256, 1)
void attn_kernel(const half_t* __restrict__ Qb,
                 const half_t* __restrict__ Kb,
                 const half_t* __restrict__ Vb,
                 half_t* __restrict__ Ob) {
  __shared__ half_t Kt[64 * 64];        // [key][d]
  __shared__ half_t Vt[64 * 64];        // transposed [d][key]
  __shared__ half_t Pt[8 * 16 * 64];    // per-wave P slabs
  const int t = threadIdx.x, lane = t & 31, wid = t >> 5;
  const int hh = lane >> 4, l15 = lane & 15;
  const int b = blockIdx.z, h = blockIdx.y, qb = blockIdx.x;
  const size_t base = (size_t)b * NS * 512 + (size_t)h * 64;
  const int qrow0 = qb * 128 + wid * 16;

  v16h qa[2];
#pragma unroll
  for (int dc = 0; dc < 2; ++dc) {
    const half_t* qp = Qb + base + (size_t)(qrow0 + l15) * 512 + dc * 32;
    union { v16h v; v8h h2[2]; } u;
    u.h2[0] = *(const v8h*)&qp[hh * 8];
    u.h2[1] = *(const v8h*)&qp[16 + hh * 8];
    qa[dc] = u.v;
  }

  v8f ofr[4];
  v8f zz = {};
#pragma unroll
  for (int i = 0; i < 4; ++i) ofr[i] = zz;
  float mrow[8], lrow[8];
#pragma unroll
  for (int r = 0; r < 8; ++r) { mrow[r] = -1e30f; lrow[r] = 0.f; }

  half_t* Pw = &Pt[wid * 16 * 64];
  const float scale = 0.125f;           // 1/sqrt(64)

  for (int kb = 0; kb < NS / 64; ++kb) {
    __syncthreads();
#ifdef USE_TDM
    // TDM: one wave issues a 2D tile load (64 rows x 64 halves, row stride
    // 512 elements) straight into Kt; tracked with TENSORcnt.
    if (wid == 0) {
      const unsigned lds_off =
          (unsigned)(size_t)(AS3 half_t*)Kt;
      const unsigned long long ga =
          (unsigned long long)(const void*)(Kb + base + (size_t)(kb * 64) * 512);
      u32x4 g0;
      g0[0] = 1u;                                    // count=1, user D#
      g0[1] = lds_off;                               // lds_addr
      g0[2] = (unsigned)(ga & 0xffffffffu);          // global_addr[31:0]
      g0[3] = (unsigned)((ga >> 32) & 0x1ffffffu) |  // global_addr[56:32]
              (2u << 30);                            // type=2 (image)
      i32x8 g1;
      g1[0] = 1 << 16;                               // data_size=2B, mask=0
      g1[1] = 64 << 16;                              // tensor_dim0 = 64
      g1[2] = 1024 << 16;                            // tensor_dim1 = 1024
      g1[3] = 64 << 16;                              // tile_dim0 = 64
      g1[4] = 64;                                    // tile_dim1 = 64
      g1[5] = 512;                                   // tensor_dim0_stride
      g1[6] = 0; g1[7] = 0;
      i32x4 gz = {0, 0, 0, 0};
#if __clang_major__ >= 23
      i32x8 gz8 = {0, 0, 0, 0, 0, 0, 0, 0};
      __builtin_amdgcn_tensor_load_to_lds(g0, g1, gz, gz, gz8, 0);
#else
      __builtin_amdgcn_tensor_load_to_lds(g0, g1, gz, gz, 0);
#endif
      __builtin_amdgcn_s_wait_tensorcnt(0);
    }
#endif
#pragma unroll
    for (int c = 0; c < 2; ++c) {       // stage V^T (and K when no TDM)
      int id = t + c * 256;
      int key = id >> 3, d8 = (id & 7) * 8;
#ifndef USE_TDM
      copy16_to_lds(&Kt[key * 64 + d8],
                    &Kb[base + (size_t)(kb * 64 + key) * 512 + d8]);
#endif
      v8h vv = *(const v8h*)&Vb[base + (size_t)(kb * 64 + key) * 512 + d8];
#pragma unroll
      for (int j = 0; j < 8; ++j) Vt[(d8 + j) * 64 + key] = vv[j];
    }
    async_join();
    __syncthreads();

    v8f sfr[4];                          // scores: Q @ K^T, 16x64 per wave
#pragma unroll
    for (int nb = 0; nb < 4; ++nb) {
      v8f z = zz;
#pragma unroll
      for (int dc = 0; dc < 2; ++dc) {
        union { v16h v; v8h h2[2]; } u;
        int key = nb * 16 + l15;
        u.h2[0] = *(const v8h*)&Kt[key * 64 + dc * 32 + hh * 8];
        u.h2[1] = *(const v8h*)&Kt[key * 64 + dc * 32 + 16 + hh * 8];
        z = __builtin_amdgcn_wmma_f32_16x16x32_f16(false, qa[dc], false, u.v,
                                                   (short)0, z, false, false);
      }
      sfr[nb] = z;
    }

    // online softmax per D-frag row (16-lane reductions within half-wave)
#pragma unroll
    for (int r = 0; r < 8; ++r) {
      float s0 = sfr[0][r] * scale, s1 = sfr[1][r] * scale;
      float s2 = sfr[2][r] * scale, s3 = sfr[3][r] * scale;
      float mx = fmaxf(fmaxf(s0, s1), fmaxf(s2, s3));
#pragma unroll
      for (int m = 8; m >= 1; m >>= 1) mx = fmaxf(mx, __shfl_xor(mx, m, 32));
      const float mnew = fmaxf(mrow[r], mx);
      const float corr = __expf(mrow[r] - mnew);
      const float p0 = __expf(s0 - mnew), p1 = __expf(s1 - mnew);
      const float p2 = __expf(s2 - mnew), p3 = __expf(s3 - mnew);
      float ps = p0 + p1 + p2 + p3;
#pragma unroll
      for (int m = 8; m >= 1; m >>= 1) ps += __shfl_xor(ps, m, 32);
      lrow[r] = lrow[r] * corr + ps;
      mrow[r] = mnew;
#pragma unroll
      for (int f = 0; f < 4; ++f) ofr[f][r] *= corr;
      const int mr = r + 8 * hh;        // D-frag row -> P slab row
      Pw[mr * 64 +  0 + l15] = (half_t)p0;
      Pw[mr * 64 + 16 + l15] = (half_t)p1;
      Pw[mr * 64 + 32 + l15] = (half_t)p2;
      Pw[mr * 64 + 48 + l15] = (half_t)p3;
    }

    v16h pa[2];                          // P as A frags (per-wave slab)
#pragma unroll
    for (int kc = 0; kc < 2; ++kc) {
      union { v16h v; v8h h2[2]; } u;
      u.h2[0] = *(const v8h*)&Pw[l15 * 64 + kc * 32 + hh * 8];
      u.h2[1] = *(const v8h*)&Pw[l15 * 64 + kc * 32 + 16 + hh * 8];
      pa[kc] = u.v;
    }
#pragma unroll
    for (int db = 0; db < 4; ++db) {     // O += P @ V
#pragma unroll
      for (int kc = 0; kc < 2; ++kc) {
        union { v16h v; v8h h2[2]; } u;
        int d = db * 16 + l15;
        u.h2[0] = *(const v8h*)&Vt[d * 64 + kc * 32 + hh * 8];
        u.h2[1] = *(const v8h*)&Vt[d * 64 + kc * 32 + 16 + hh * 8];
        ofr[db] = __builtin_amdgcn_wmma_f32_16x16x32_f16(
            false, pa[kc], false, u.v, (short)0, ofr[db], false, false);
      }
    }
  }

#pragma unroll
  for (int r = 0; r < 8; ++r) {
    const float inv = 1.0f / lrow[r];
    const int row = qrow0 + r + 8 * hh;
#pragma unroll
    for (int db = 0; db < 4; ++db)
      Ob[base + (size_t)row * 512 + db * 16 + l15] =
          (half_t)(ofr[db][r] * inv);
  }
}

// ---------------------------------------------------------------- driver
extern "C" void kernel_launch(void* const* d_in, const int* in_sizes, int n_in,
                              void* d_out, int out_size, void* d_ws,
                              size_t ws_size, hipStream_t stream) {
  (void)in_sizes; (void)n_in; (void)out_size; (void)ws_size;
  const float* inputs    = (const float*)d_in[0];
  const float* rms_scale = (const float*)d_in[1];
  const float* q_w       = (const float*)d_in[2];
  const float* k_w       = (const float*)d_in[3];
  const float* v_w       = (const float*)d_in[4];
  const float* Wq        = (const float*)d_in[5];
  const float* bq        = (const float*)d_in[6];
  const float* Wk        = (const float*)d_in[7];
  const float* bk        = (const float*)d_in[8];
  const float* Wv        = (const float*)d_in[9];
  const float* bv        = (const float*)d_in[10];
  const float* Wo        = (const float*)d_in[11];
  const float* bo        = (const float*)d_in[12];
  const float* gamma     = (const float*)d_in[13];
  float* out = (float*)d_out;

  char* ws = (char*)d_ws;
  size_t off = 0;
  auto alloc = [&](size_t bytes) -> void* {
    void* p = ws + off;
    off = (off + bytes + 255) & ~(size_t)255;
    return p;
  };
  half_t* x16   = (half_t*)alloc((size_t)BHW * NC * 2);
  half_t* qbuf  = (half_t*)alloc((size_t)BHW * NC * 2);
  half_t* kbuf  = (half_t*)alloc((size_t)BHW * NC * 2);
  half_t* vbuf  = (half_t*)alloc((size_t)BHW * NC * 2);
  half_t* ctx   = (half_t*)alloc((size_t)BHW * NC * 2);
  half_t* kc    = (half_t*)alloc((size_t)BHW * 64 * 2);
  half_t* vc    = (half_t*)alloc((size_t)BHW * 64 * 2);
  half_t* wcomb = (half_t*)alloc((size_t)512 * 512 * 2);
  half_t* wkred = (half_t*)alloc((size_t)64 * 512 * 2);
  half_t* wvred = (half_t*)alloc((size_t)64 * 512 * 2);
  half_t* wo16  = (half_t*)alloc((size_t)512 * 512 * 2);

  rmsnorm_kernel<<<BHW / 8, 256, 0, stream>>>(inputs, rms_scale, x16);
  conv_kv_kernel<<<(BHW * 64) / 256, 256, 0, stream>>>(x16, k_w, v_w, kc, vc);
  wcomb_kernel<<<(512 * 512) / 256, 256, 0, stream>>>(q_w, Wq, wcomb);
  wred_kernel<<<(64 * 512) / 256, 256, 0, stream>>>(Wk, Wv, wkred, wvred);
  cvt16_kernel<<<(512 * 512) / 256, 256, 0, stream>>>(Wo, wo16, 512 * 512);

  dim3 gg(512 / 128, BHW / 128);
  gemm_kernel<0><<<gg, 256, 0, stream>>>(x16, wcomb, bq, 512, qbuf,
                                         nullptr, nullptr, nullptr);
  gemm_kernel<0><<<gg, 256, 0, stream>>>(kc, wkred, bk, 64, kbuf,
                                         nullptr, nullptr, nullptr);
  gemm_kernel<0><<<gg, 256, 0, stream>>>(vc, wvred, bv, 64, vbuf,
                                         nullptr, nullptr, nullptr);

  dim3 ga(NS / 128, HEADS, NB);
  attn_kernel<<<ga, 256, 0, stream>>>(qbuf, kbuf, vbuf, ctx);

  gemm_kernel<1><<<gg, 256, 0, stream>>>(ctx, wo16, bo, 512, nullptr, out,
                                         gamma, inputs);
}
